// minGRUCell_75531294867629
// MI455X (gfx1250) — compile-verified, matrix operand order checked by reading, not run on hardware
//
#include <hip/hip_runtime.h>
#include <math.h>

// ---------------- problem constants ----------------
#define BB    8
#define TT    4096
#define DD    512
#define HH    512
#define NROWS (BB * TT)        // 32768 rows of the GEMM
#define NC    64               // scan chunks
#define CLEN  (TT / NC)        // 64 steps per chunk

// ---------------- types ----------------
typedef __bf16 bf16_t;
typedef bf16_t v16bf __attribute__((ext_vector_type(16)));
typedef bf16_t v8bf  __attribute__((ext_vector_type(8)));
typedef bf16_t v4bf  __attribute__((ext_vector_type(4)));
typedef float  v8f   __attribute__((ext_vector_type(8)));

// ---------------- f32 -> bf16 conversion (memory-bound prologue) ----------------
__global__ __launch_bounds__(256)
void cvt_f32_to_bf16(const float* __restrict__ src, bf16_t* __restrict__ dst, int n) {
    int i = (blockIdx.x * blockDim.x + threadIdx.x) * 4;
    if (i < n) {
        float4 v = *(const float4*)(src + i);
        v4bf o;
        o[0] = (bf16_t)v.x; o[1] = (bf16_t)v.y;
        o[2] = (bf16_t)v.z; o[3] = (bf16_t)v.w;
        *(v4bf*)(dst + i) = o;
    }
}

// ---------------- WMMA fragment load ----------------
// 16-bit A/B layout (ISA 7.12.2): lane<16 holds K = {0..7, 16..23},
// lane>=16 holds K = {8..15, 24..31} of its row. Two aligned 16B loads.
__device__ inline v16bf load_frag(const bf16_t* __restrict__ p) {
    v8bf lo = *(const v8bf*)(p);
    v8bf hi = *(const v8bf*)(p + 16);
    v16bf f;
#pragma unroll
    for (int i = 0; i < 8; ++i) { f[i] = lo[i]; f[i + 8] = hi[i]; }
    return f;
}

// ---------------- fused dual-GEMM + activations ----------------
// Computes z = sigmoid(x Wz^T + bz), ht = tanh(x Wh^T + bh),
// stores a = 1-z into a_out (== d_out) and b = z*ht into b_out (ws).
// Block: 128 threads = 4 waves in 2x2; each wave does a 32x32 tile pair.
__global__ __launch_bounds__(128)
void gemm_gate_kernel(const bf16_t* __restrict__ xb,
                      const bf16_t* __restrict__ wzb,
                      const bf16_t* __restrict__ whb,
                      const float*  __restrict__ bz,
                      const float*  __restrict__ bh,
                      float* __restrict__ a_out,
                      float* __restrict__ b_out) {
    const int lane = threadIdx.x & 31;
    const int wave = threadIdx.x >> 5;
    const int wm   = wave >> 1;
    const int wn   = wave & 1;
    const int m0   = blockIdx.y * 64 + wm * 32;   // row block (n = b*T + t)
    const int n0   = blockIdx.x * 64 + wn * 32;   // col block (h)
    const int r    = lane & 15;
    const int koff = (lane < 16) ? 0 : 8;

    v8f accZ[2][2] = {}, accH[2][2] = {};

    const bf16_t* aP[2];
    const bf16_t* zP[2];
    const bf16_t* hP[2];
#pragma unroll
    for (int i = 0; i < 2; ++i) {
        aP[i] = xb  + (size_t)(m0 + i * 16 + r) * DD + koff;
        zP[i] = wzb + (size_t)(n0 + i * 16 + r) * DD + koff;
        hP[i] = whb + (size_t)(n0 + i * 16 + r) * DD + koff;
    }

#pragma unroll 4
    for (int k0 = 0; k0 < DD; k0 += 32) {
        v16bf aF[2], zF[2], hF[2];
#pragma unroll
        for (int i = 0; i < 2; ++i) {
            aF[i] = load_frag(aP[i] + k0);
            zF[i] = load_frag(zP[i] + k0);
            hF[i] = load_frag(hP[i] + k0);
        }
#pragma unroll
        for (int mt = 0; mt < 2; ++mt)
#pragma unroll
            for (int nt = 0; nt < 2; ++nt) {
                accZ[mt][nt] = __builtin_amdgcn_wmma_f32_16x16x32_bf16(
                    false, aF[mt], false, zF[nt], (short)0, accZ[mt][nt], false, false);
                accH[mt][nt] = __builtin_amdgcn_wmma_f32_16x16x32_bf16(
                    false, aF[mt], false, hF[nt], (short)0, accH[mt][nt], false, false);
            }
    }

    // Epilogue: C/D layout — VGPR e, lanes 0-15: M=e, N=lane; lanes 16-31: M=e+8.
    const int rowAdd = (lane < 16) ? 0 : 8;
#pragma unroll
    for (int nt = 0; nt < 2; ++nt) {
        const int col = n0 + nt * 16 + r;
        const float bzv = bz[col];
        const float bhv = bh[col];
#pragma unroll
        for (int mt = 0; mt < 2; ++mt) {
#pragma unroll
            for (int e = 0; e < 8; ++e) {
                const int row = m0 + mt * 16 + e + rowAdd;
                const float zarg = accZ[mt][nt][e] + bzv;
                const float z    = 1.0f / (1.0f + __expf(-zarg));
                const float ht   = tanhf(accH[mt][nt][e] + bhv);
                const size_t idx = (size_t)row * HH + col;
                a_out[idx] = 1.0f - z;   // multiplicative coefficient
                b_out[idx] = z * ht;     // additive term
            }
        }
    }
}

// ---------------- pass 1: per-chunk composition (A, B) ----------------
__global__ __launch_bounds__(256)
void scan_chunk_reduce(const float* __restrict__ a_arr,
                       const float* __restrict__ b_arr,
                       float* __restrict__ Ac, float* __restrict__ Bc) {
    const int gid = blockIdx.x * blockDim.x + threadIdx.x;  // B*NC*H threads
    const int h = gid & (HH - 1);
    const int c = (gid >> 9) & (NC - 1);
    const int b = gid >> 15;
    size_t base = ((size_t)b * TT + (size_t)c * CLEN) * HH + h;
    float A = 1.0f, Bv = 0.0f;
#pragma unroll 4
    for (int t = 0; t < CLEN; ++t) {
        const float at = a_arr[base];
        const float bt = b_arr[base];
        Bv = fmaf(at, Bv, bt);
        A  = A * at;
        base += HH;
    }
    const size_t ci = ((size_t)b * NC + c) * HH + h;
    Ac[ci] = A;
    Bc[ci] = Bv;
}

// ---------------- pass 2: carry scan over chunks (tiny) ----------------
__global__ __launch_bounds__(256)
void scan_carry(const float* __restrict__ Ac, const float* __restrict__ Bc,
                const float* __restrict__ h0, float* __restrict__ carry) {
    const int gid = blockIdx.x * blockDim.x + threadIdx.x;  // B*H threads
    const int h = gid & (HH - 1);
    const int b = gid >> 9;
    float hv = h0[(size_t)b * HH + h];
#pragma unroll 8
    for (int c = 0; c < NC; ++c) {
        const size_t ci = ((size_t)b * NC + c) * HH + h;
        carry[ci] = hv;                        // h entering chunk c
        hv = fmaf(Ac[ci], hv, Bc[ci]);
    }
}

// ---------------- pass 3: replay chunks, write outputs ----------------
__global__ __launch_bounds__(256)
void scan_apply(const float* __restrict__ carry,
                const float* __restrict__ b_arr,
                float* __restrict__ out /* holds a, overwritten with h */) {
    const int gid = blockIdx.x * blockDim.x + threadIdx.x;  // B*NC*H threads
    const int h = gid & (HH - 1);
    const int c = (gid >> 9) & (NC - 1);
    const int b = gid >> 15;
    float hv = carry[((size_t)b * NC + c) * HH + h];
    size_t base = ((size_t)b * TT + (size_t)c * CLEN) * HH + h;
#pragma unroll 4
    for (int t = 0; t < CLEN; ++t) {
        const float at = out[base];      // a stored in d_out (read-before-write)
        const float bt = b_arr[base];
        hv = fmaf(at, hv, bt);
        out[base] = hv;
        base += HH;
    }
}

// ---------------- launcher ----------------
extern "C" void kernel_launch(void* const* d_in, const int* in_sizes, int n_in,
                              void* d_out, int out_size, void* d_ws, size_t ws_size,
                              hipStream_t stream) {
    const float* x  = (const float*)d_in[0];   // (B,T,D)
    const float* h0 = (const float*)d_in[1];   // (B,1,H)
    const float* Wz = (const float*)d_in[2];   // (H,D)
    const float* bz = (const float*)d_in[3];   // (H)
    const float* Wh = (const float*)d_in[4];   // (H,D)
    const float* bh = (const float*)d_in[5];   // (H)
    float* out = (float*)d_out;                // (B,T,H)

    char* ws = (char*)d_ws;
    bf16_t* xb    = (bf16_t*)(ws);                                   // 32 MB
    bf16_t* wzb   = (bf16_t*)(ws + (size_t)33554432);                // 512 KB
    bf16_t* whb   = (bf16_t*)(ws + (size_t)34078720);                // 512 KB
    float*  bvals = (float*) (ws + (size_t)34603008);                // 64 MB
    float*  Ac    = (float*) (ws + (size_t)101711872);               // 1 MB
    float*  Bc    = (float*) (ws + (size_t)102760448);               // 1 MB
    float*  carry = (float*) (ws + (size_t)103809024);               // 1 MB

    // 1) convert x, Wz, Wh to bf16
    cvt_f32_to_bf16<<<(NROWS * DD / 4 + 255) / 256, 256, 0, stream>>>(x, xb, NROWS * DD);
    cvt_f32_to_bf16<<<(HH * DD / 4 + 255) / 256, 256, 0, stream>>>(Wz, wzb, HH * DD);
    cvt_f32_to_bf16<<<(HH * DD / 4 + 255) / 256, 256, 0, stream>>>(Wh, whb, HH * DD);

    // 2) fused dual GEMM + gates: a -> d_out, b -> ws
    dim3 grid(HH / 64, NROWS / 64);   // (8, 512)
    gemm_gate_kernel<<<grid, 128, 0, stream>>>(xb, wzb, whb, bz, bh, out, bvals);

    // 3) chunked linear-recurrence scan
    scan_chunk_reduce<<<(BB * NC * HH) / 256, 256, 0, stream>>>(out, bvals, Ac, Bc);
    scan_carry<<<(BB * HH) / 256, 256, 0, stream>>>(Ac, Bc, h0, carry);
    scan_apply<<<(BB * NC * HH) / 256, 256, 0, stream>>>(carry, bvals, out);
}
